// MultiHeadAttention_21655225107027
// MI455X (gfx1250) — matrix operand order repaired and checked by reading, not compiled, not run on hardware
//
#include <hip/hip_runtime.h>

// MHA forward: B=2, N=2048, D=1024, H=16, DK=64 on gfx1250 (wave32, WMMA).
// Pipeline:
//   1) convert inputs + weights to bf16 once (pure bandwidth pass)
//   2) 3x bf16 WMMA projections (32 tokens x 64 feats per wave, 8 WMMA/k-step)
//   3) flash attention: 4-wave workgroups share double-buffered async-staged
//      K/V LDS tiles; online softmax; P C->A layout via per-wave LDS
//   4) output projection (bf16 WMMA, f32 out)
// Workspace (bf16 elements, 64 MB total):
//   Xq/Xk/Xv: 3 x 4M   | Wq/Wk/Wv/Wo: 4 x 1M | Qh/Kh: 2 x 4M [B,H,N,DK]
//   Vt: 4M [B,H,DK,N]  | CTX: 4M [B,N,D]

typedef __attribute__((ext_vector_type(16))) __bf16 bf16x16;
typedef __attribute__((ext_vector_type(8)))  __bf16 bf16x8;
typedef __attribute__((ext_vector_type(8)))  float  f32x8;
typedef __attribute__((ext_vector_type(4)))  int    v4i;

static constexpr int B_ = 2, N_ = 2048, D_ = 1024, H_ = 16, DK_ = 64;

#if defined(__AMDGCN__) && \
    __has_builtin(__builtin_amdgcn_global_load_async_to_lds_b128) && \
    __has_builtin(__builtin_amdgcn_s_wait_asynccnt)
#define HAS_ASYNC 1
#else
#define HAS_ASYNC 0
#endif

__device__ __forceinline__ f32x8 wmma_bf16(bf16x16 a, bf16x16 b, f32x8 c) {
    return __builtin_amdgcn_wmma_f32_16x16x32_bf16(false, a, false, b, (short)0, c,
                                                   false, false);
}

__device__ __forceinline__ bf16x16 cat16(bf16x8 lo, bf16x8 hi) {
    bf16x16 r;
#pragma unroll
    for (int i = 0; i < 8; ++i) { r[i] = lo[i]; r[8 + i] = hi[i]; }
    return r;
}

// 16-byte global->LDS copy: async on gfx1250, synchronous fallback elsewhere.
__device__ __forceinline__ void cp16B(__bf16* lds_dst, const __bf16* gsrc) {
#if HAS_ASYNC
    __builtin_amdgcn_global_load_async_to_lds_b128(
        (__attribute__((address_space(1))) v4i*)gsrc,
        (__attribute__((address_space(3))) v4i*)lds_dst, 0, 0);
#else
    *(uint4*)(void*)lds_dst = *(const uint4*)(const void*)gsrc;
#endif
}

#if HAS_ASYNC
#define WAIT_ASYNC(n) __builtin_amdgcn_s_wait_asynccnt(n)
#else
#define WAIT_ASYNC(n) ((void)0)
#endif

// ---------------------------------------------------------------------------
// f32 -> bf16 bulk convert (8 elements/thread)
// ---------------------------------------------------------------------------
__global__ void __launch_bounds__(256)
cvt_bf16_kernel(const float* __restrict__ src, __bf16* __restrict__ dst, int n8)
{
    const int i = blockIdx.x * 256 + threadIdx.x;
    if (i < n8) {
        const float* p = src + (size_t)i * 8;
        bf16x8 r;
#pragma unroll
        for (int j = 0; j < 8; ++j) r[j] = (__bf16)p[j];
        *(bf16x8*)(dst + (size_t)i * 8) = r;
    }
}

// ---------------------------------------------------------------------------
// Projection: out = X @ W^T + b. X bf16 [4096,1024], W bf16 [1024,1024].
// One wave: 32 tokens x 64 features, 8 WMMAs per k-step.
// transposed==0 -> [B,H,N,DK] ; transposed==1 -> [B,H,DK,N]
// ---------------------------------------------------------------------------
__global__ void __launch_bounds__(32)
proj32_kernel(const __bf16* __restrict__ X, const __bf16* __restrict__ W,
              const float* __restrict__ bias, __bf16* __restrict__ out, int transposed)
{
    const int lane = threadIdx.x & 31;
    const int col  = lane & 15;
    const int half = lane >> 4;
    const int tokBase = blockIdx.x * 32;
    const int nBase   = blockIdx.y * 64;

    const __bf16* xrow0 = X + (size_t)(tokBase + col) * D_;
    const __bf16* xrow1 = X + (size_t)(tokBase + 16 + col) * D_;
    f32x8 acc[2][4] = {};

    for (int k = 0; k < D_; k += 32) {
        bf16x16 A0 = cat16(*(const bf16x8*)(xrow0 + k + half * 16),
                           *(const bf16x8*)(xrow0 + k + half * 16 + 8));
        bf16x16 A1 = cat16(*(const bf16x8*)(xrow1 + k + half * 16),
                           *(const bf16x8*)(xrow1 + k + half * 16 + 8));
        __builtin_prefetch(xrow0 + k + 256, 0, 0);
        __builtin_prefetch(xrow1 + k + 256, 0, 0);
#pragma unroll
        for (int t = 0; t < 4; ++t) {
            const __bf16* wrow = W + (size_t)(nBase + t * 16 + col) * D_ + k + half * 16;
            bf16x16 Bv = *(const bf16x16*)wrow;
            acc[0][t] = wmma_bf16(A0, Bv, acc[0][t]);
            acc[1][t] = wmma_bf16(A1, Bv, acc[1][t]);
        }
    }

#pragma unroll
    for (int u = 0; u < 2; ++u) {
#pragma unroll
        for (int t = 0; t < 4; ++t) {
            const int nf = nBase + t * 16 + col;
            const float bv = bias[nf];
            const int h = nf >> 6, d = nf & (DK_ - 1);
            if (!transposed) {
#pragma unroll
                for (int r = 0; r < 8; ++r) {
                    const int tok = tokBase + u * 16 + r + 8 * half;
                    const int b = tok >> 11, tt = tok & (N_ - 1);
                    out[((size_t)(b * H_ + h) * N_ + tt) * DK_ + d] =
                        (__bf16)(acc[u][t][r] + bv);
                }
            } else {
                const int tok0 = tokBase + u * 16 + 8 * half;
                const int b = tok0 >> 11, tt0 = tok0 & (N_ - 1);
                __bf16* dst = out + ((size_t)(b * H_ + h) * DK_ + d) * N_ + tt0;
#pragma unroll
                for (int r = 0; r < 8; ++r) dst[r] = (__bf16)(acc[u][t][r] + bv);
            }
        }
    }
}

// ---------------------------------------------------------------------------
// Flash attention. Block = 128 threads (4 waves) = 64 consecutive queries of
// one (b,h). K/V tiles (32 keys) staged into double-buffered LDS with async
// global->LDS loads, shared by all 4 waves. Online softmax per wave.
// ---------------------------------------------------------------------------
__global__ void __launch_bounds__(128)
attn_kernel(const __bf16* __restrict__ Qh, const __bf16* __restrict__ Kh,
            const __bf16* __restrict__ Vt, __bf16* __restrict__ ctx)
{
    __shared__ alignas(32) __bf16 Kbuf[2][32 * 64];   // [buf][j_local*64 + d]
    __shared__ alignas(32) __bf16 Vbuf[2][64 * 32];   // [buf][d*32 + j_local]
    __shared__ alignas(32) __bf16 Pbuf[4][16 * 34];   // per-wave P, padded rows

    const int tid  = threadIdx.x;
    const int w    = tid >> 5;
    const int lane = tid & 31;
    const int col  = lane & 15;
    const int half = lane >> 4;
    const int qblock = blockIdx.x * 64;
    const int h  = blockIdx.y;
    const int b  = blockIdx.z;
    const int bh = b * H_ + h;
    const int qb   = qblock + w * 16;
    const int qEnd = qb + 15;
    const float sscale = 0.125f;            // 1/sqrt(DK)

    // Q A-operands (d=0..31, 32..63) kept in registers
    const __bf16* qrow = Qh + ((size_t)bh * N_ + (qb + col)) * DK_;
    bf16x16 Qa[2];
#pragma unroll
    for (int s = 0; s < 2; ++s)
        Qa[s] = cat16(*(const bf16x8*)(qrow + s * 32 + half * 16),
                      *(const bf16x8*)(qrow + s * 32 + half * 16 + 8));

    float runmax[8], runsum[8];
#pragma unroll
    for (int r = 0; r < 8; ++r) { runmax[r] = -1e30f; runsum[r] = 0.0f; }
    f32x8 acc[4] = {};

    const int nsteps = qblock / 32 + 2;     // keys 0 .. qblock+63

    // stage a 32-key K/V tile into LDS buffer (4 x 16B chunks per thread)
    auto stage = [&](int jb, int buf) {
#pragma unroll
        for (int i = 0; i < 2; ++i) {
            const int c = tid + i * 128;
            {   // K tile: 32 rows x 64 bf16 = 256 x 16B chunks
                const int row = c >> 3, off = (c & 7) * 8;
                cp16B(&Kbuf[buf][row * 64 + off],
                      Kh + ((size_t)bh * N_ + jb + row) * DK_ + off);
            }
            {   // V tile: 64 rows x 32 bf16 = 256 x 16B chunks
                const int row = c >> 2, off = (c & 3) * 8;
                cp16B(&Vbuf[buf][row * 32 + off],
                      Vt + ((size_t)bh * DK_ + row) * N_ + jb + off);
            }
        }
    };

    stage(0, 0);
    int buf = 0;

    for (int step = 0; step < nsteps; ++step) {
        const int jb0 = step * 32;
        const bool more = (step + 1) < nsteps;
        if (more) stage(jb0 + 32, buf ^ 1);          // prefetch next tile
        if (more) WAIT_ASYNC(4); else WAIT_ASYNC(0); // current tile landed
        __syncthreads();

        if (jb0 <= qEnd) {
            const bool t1v = (jb0 + 16) <= qEnd;

            // --- S = Q K^T from LDS K tile ---
            f32x8 s0 = {}, s1 = {};
#pragma unroll
            for (int s = 0; s < 2; ++s) {
                bf16x16 Bv = *(const bf16x16*)&Kbuf[buf][col * 64 + s * 32 + half * 16];
                s0 = wmma_bf16(Qa[s], Bv, s0);
            }
            if (t1v) {
#pragma unroll
                for (int s = 0; s < 2; ++s) {
                    bf16x16 Bv =
                        *(const bf16x16*)&Kbuf[buf][(16 + col) * 64 + s * 32 + half * 16];
                    s1 = wmma_bf16(Qa[s], Bv, s1);
                }
            }

            // --- causal mask + online softmax (row = qb + r + 8*half) ---
#pragma unroll
            for (int r = 0; r < 8; ++r) {
                const int m = qb + r + 8 * half;
                float v0 = s0[r] * sscale;
                if (jb0 + col > m) v0 = -1e30f;
                float v1 = t1v ? s1[r] * sscale : -1e30f;
                if (jb0 + 16 + col > m) v1 = -1e30f;

                float mx = fmaxf(v0, v1);
                mx = fmaxf(mx, __shfl_xor(mx, 1));
                mx = fmaxf(mx, __shfl_xor(mx, 2));
                mx = fmaxf(mx, __shfl_xor(mx, 4));
                mx = fmaxf(mx, __shfl_xor(mx, 8));

                const float nm = fmaxf(runmax[r], mx);
                const float rescale = __expf(runmax[r] - nm);
                runmax[r] = nm;

                const float p0 = __expf(v0 - nm);
                const float p1 = __expf(v1 - nm);
                float rs = p0 + p1;
                rs += __shfl_xor(rs, 1);
                rs += __shfl_xor(rs, 2);
                rs += __shfl_xor(rs, 4);
                rs += __shfl_xor(rs, 8);
                runsum[r] = runsum[r] * rescale + rs;

#pragma unroll
                for (int t = 0; t < 4; ++t) acc[t][r] *= rescale;

                Pbuf[w][(r + 8 * half) * 34 + col]      = (__bf16)p0;
                Pbuf[w][(r + 8 * half) * 34 + 16 + col] = (__bf16)p1;
            }
            // per-wave LDS: DS ops are in-order within a wave, no barrier needed

            // --- P in A-layout (16x32) from per-wave LDS ---
            bf16x16 Pa;
#pragma unroll
            for (int v = 0; v < 8; ++v) {
                const int K0 = half * 16 + v * 2;
                Pa[2 * v]     = Pbuf[w][col * 34 + K0];
                Pa[2 * v + 1] = Pbuf[w][col * 34 + K0 + 1];
            }

            // --- ctx += P V from LDS V tile ---
#pragma unroll
            for (int t = 0; t < 4; ++t) {
                bf16x16 Bv =
                    *(const bf16x16*)&Vbuf[buf][(t * 16 + col) * 32 + half * 16];
                acc[t] = wmma_bf16(Pa, Bv, acc[t]);
            }
        }

        __syncthreads();   // everyone done with buf before it is re-staged
        buf ^= 1;
    }

    // --- normalize, store ctx bf16 [B,N,D] ---
#pragma unroll
    for (int t = 0; t < 4; ++t) {
#pragma unroll
        for (int r = 0; r < 8; ++r) {
            const float o = acc[t][r] / runsum[r];
            const int tok = qb + r + 8 * half;
            ctx[((size_t)b * N_ + tok) * D_ + h * DK_ + t * 16 + col] = (__bf16)o;
        }
    }
}

// ---------------------------------------------------------------------------
// Output projection: out = ctx @ Wo^T + bo (bf16 operands, f32 out)
// ---------------------------------------------------------------------------
__global__ void __launch_bounds__(32)
oproj32_kernel(const __bf16* __restrict__ X, const __bf16* __restrict__ W,
               const float* __restrict__ bias, float* __restrict__ out)
{
    const int lane = threadIdx.x & 31;
    const int col  = lane & 15;
    const int half = lane >> 4;
    const int tokBase = blockIdx.x * 32;
    const int nBase   = blockIdx.y * 64;

    const __bf16* xrow0 = X + (size_t)(tokBase + col) * D_;
    const __bf16* xrow1 = X + (size_t)(tokBase + 16 + col) * D_;
    f32x8 acc[2][4] = {};

    for (int k = 0; k < D_; k += 32) {
        bf16x16 A0 = cat16(*(const bf16x8*)(xrow0 + k + half * 16),
                           *(const bf16x8*)(xrow0 + k + half * 16 + 8));
        bf16x16 A1 = cat16(*(const bf16x8*)(xrow1 + k + half * 16),
                           *(const bf16x8*)(xrow1 + k + half * 16 + 8));
        __builtin_prefetch(xrow0 + k + 256, 0, 0);
#pragma unroll
        for (int t = 0; t < 4; ++t) {
            const __bf16* wrow = W + (size_t)(nBase + t * 16 + col) * D_ + k + half * 16;
            bf16x16 Bv = *(const bf16x16*)wrow;
            acc[0][t] = wmma_bf16(A0, Bv, acc[0][t]);
            acc[1][t] = wmma_bf16(A1, Bv, acc[1][t]);
        }
    }

#pragma unroll
    for (int u = 0; u < 2; ++u)
#pragma unroll
        for (int t = 0; t < 4; ++t) {
            const float bv = bias[nBase + t * 16 + col];
#pragma unroll
            for (int r = 0; r < 8; ++r) {
                const int tok = tokBase + u * 16 + r + 8 * half;
                out[(size_t)tok * D_ + nBase + t * 16 + col] = acc[u][t][r] + bv;
            }
        }
}

// ---------------------------------------------------------------------------
extern "C" void kernel_launch(void* const* d_in, const int* in_sizes, int n_in,
                              void* d_out, int out_size, void* d_ws, size_t ws_size,
                              hipStream_t stream) {
    const float* query = (const float*)d_in[0];
    const float* key   = (const float*)d_in[1];
    const float* value = (const float*)d_in[2];
    // d_in[3] = causal mask, computed analytically instead
    const float* Wq = (const float*)d_in[4];
    const float* bq = (const float*)d_in[5];
    const float* Wk = (const float*)d_in[6];
    const float* bk = (const float*)d_in[7];
    const float* Wv = (const float*)d_in[8];
    const float* bv = (const float*)d_in[9];
    const float* Wo = (const float*)d_in[10];
    const float* bo = (const float*)d_in[11];
    float* out = (float*)d_out;

    char* ws = (char*)d_ws;
    size_t off = 0;   // in bf16 elements
    auto take = [&](size_t n) { __bf16* p = (__bf16*)(ws + off * 2); off += n; return p; };
    const size_t ACT = (size_t)B_ * N_ * D_;   // 4M
    const size_t WEL = (size_t)D_ * D_;        // 1M
    __bf16* Xq  = take(ACT);
    __bf16* Xk  = take(ACT);
    __bf16* Xv  = take(ACT);
    __bf16* Wqb = take(WEL);
    __bf16* Wkb = take(WEL);
    __bf16* Wvb = take(WEL);
    __bf16* Wob = take(WEL);
    __bf16* Qh  = take(ACT);
    __bf16* Kh  = take(ACT);
    __bf16* Vt  = take(ACT);
    __bf16* CTX = take(ACT);

    const dim3 blk32(32);
    // --- 1) bf16 conversion passes ---
    {
        const int nA8 = (int)(ACT / 8), nW8 = (int)(WEL / 8);
        const dim3 b(256);
        hipLaunchKernelGGL(cvt_bf16_kernel, dim3((nA8 + 255) / 256), b, 0, stream, query, Xq, nA8);
        hipLaunchKernelGGL(cvt_bf16_kernel, dim3((nA8 + 255) / 256), b, 0, stream, key,   Xk, nA8);
        hipLaunchKernelGGL(cvt_bf16_kernel, dim3((nA8 + 255) / 256), b, 0, stream, value, Xv, nA8);
        hipLaunchKernelGGL(cvt_bf16_kernel, dim3((nW8 + 255) / 256), b, 0, stream, Wq, Wqb, nW8);
        hipLaunchKernelGGL(cvt_bf16_kernel, dim3((nW8 + 255) / 256), b, 0, stream, Wk, Wkb, nW8);
        hipLaunchKernelGGL(cvt_bf16_kernel, dim3((nW8 + 255) / 256), b, 0, stream, Wv, Wvb, nW8);
        hipLaunchKernelGGL(cvt_bf16_kernel, dim3((nW8 + 255) / 256), b, 0, stream, Wo, Wob, nW8);
    }

    // --- 2) projections ---
    const dim3 gProj((B_ * N_) / 32, D_ / 64);   // (128,16)
    hipLaunchKernelGGL(proj32_kernel, gProj, blk32, 0, stream, Xq, Wqb, bq, Qh, 0);
    hipLaunchKernelGGL(proj32_kernel, gProj, blk32, 0, stream, Xk, Wkb, bk, Kh, 0);
    hipLaunchKernelGGL(proj32_kernel, gProj, blk32, 0, stream, Xv, Wvb, bv, Vt, 1);

    // --- 3) attention ---
    const dim3 gAttn(N_ / 64, H_, B_);           // (32,16,2)
    hipLaunchKernelGGL(attn_kernel, gAttn, dim3(128), 0, stream, Qh, Kh, Vt, CTX);

    // --- 4) output projection ---
    hipLaunchKernelGGL(oproj32_kernel, gProj, blk32, 0, stream, CTX, Wob, bo, out);
}
